// Conv1d_NN_4818953307006
// MI455X (gfx1250) — matrix operand order repaired and checked by reading, not compiled, not run on hardware
//
#include <hip/hip_runtime.h>
#include <hip/hip_bf16.h>
#include <math.h>

// Problem constants (from reference)
#define BATCH 16
#define CIN   256
#define COUT  256
#define NPT   4096
#define KNN   3
#define KK_TOT (KNN * CIN)   // 768

typedef __attribute__((ext_vector_type(16))) _Float16 v16h;
typedef __attribute__((ext_vector_type(8)))  _Float16 v8h;
typedef __attribute__((ext_vector_type(8)))  float    v8f;
typedef __attribute__((ext_vector_type(4))) unsigned int u32x4;
typedef __attribute__((ext_vector_type(8))) int          i32x8;
typedef __attribute__((ext_vector_type(4))) int          i32x4;

// Workspace layout (bytes)
#define XH_BYTES  ((size_t)BATCH * NPT * CIN * 2)            // 33,554,432
#define SQ_BYTES  ((size_t)BATCH * NPT * 4)                  //    262,144
#define WH_BYTES  ((size_t)COUT * KK_TOT * 2)                //    393,216

// LDS row stride for staged B tiles: 256 f16 (512B) + 16B TDM pad = 528B.
// 528B = 132 dwords == 4 (mod 64 banks) -> consecutive rows hit distinct banks.
#define BROW_BYTES 528

#if __has_builtin(__builtin_amdgcn_tensor_load_to_lds)
#define HAVE_TDM 1
#else
#define HAVE_TDM 0
#endif

// ---------------------------------------------------------------------------
// WMMA helper: v_wmma_f32_16x16x32_f16, D = A(16x32) * B(32x16) + C(16x16)
// ---------------------------------------------------------------------------
__device__ __forceinline__ v8f wmma_f16(v16h a, v16h b, v8f c) {
  return __builtin_amdgcn_wmma_f32_16x16x32_f16(
      /*neg_a=*/false, a, /*neg_b=*/false, b,
      /*c_mod=*/(short)0, c, /*reuse_a=*/false, /*reuse_b=*/false);
}

// A operand (16x32 f16, ISA §7.12.2): lane l holds row (l&15);
// halves 0..7 -> K = kb+h, halves 8..15 -> K = kb+16+(h-8), kb=(l<16)?0:8.
__device__ __forceinline__ v16h load_a_chunk(const _Float16* base, int ld,
                                             int kc, int lane) {
  const int row = lane & 15;
  const int kb  = (lane < 16) ? 0 : 8;
  const _Float16* p = base + (size_t)row * ld + kc + kb;
  v8h lo = *(const v8h*)(p);        // 16B aligned
  v8h hi = *(const v8h*)(p + 16);   // 16B aligned
  v16h r;
#pragma unroll
  for (int h = 0; h < 8; ++h) { r[h] = lo[h]; r[h + 8] = hi[h]; }
  return r;
}

// B operand (32x16 f16): lane l holds column (l&15);
// halves 0..7 -> K = kc+kb+h, halves 8..15 -> K = kc+kb+16+(h-8), kb=(l<16)?0:8.
// Two 16B loads so only 16B alignment is required (works for global and LDS).
__device__ __forceinline__ v16h load_b_chunk(const _Float16* colrow, int kc,
                                             int lane) {
  const _Float16* p = colrow + kc + ((lane < 16) ? 0 : 8);
  v8h lo = *(const v8h*)(p);
  v8h hi = *(const v8h*)(p + 16);
  v16h r;
#pragma unroll
  for (int h = 0; h < 8; ++h) { r[h] = lo[h]; r[h + 8] = hi[h]; }
  return r;
}

// ---------------------------------------------------------------------------
// TDM: stage a 2D f16 tile (d1 rows x d0 elems, row stride in elems) into LDS.
// Padded mode inserts 4 dwords after every 128 dwords (16B per 512B row for
// the knn B tile) -> LDS row stride BROW_BYTES. D# per ISA §8.3-8.4.
// ---------------------------------------------------------------------------
#if HAVE_TDM
__device__ __forceinline__ void tdm_load_2d_f16(unsigned lds_off, const void* g,
                                                unsigned d0, unsigned d1,
                                                unsigned stride, bool pad) {
  unsigned long long ga = (unsigned long long)g;
  u32x4 g0;
  g0[0] = 1u;                                            // count=1, user mode
  g0[1] = lds_off;                                       // lds_addr (bytes)
  g0[2] = (unsigned)ga;                                  // global_addr[31:0]
  g0[3] = (unsigned)((ga >> 32) & 0x01FFFFFFull) | 0x80000000u;  // type=2
  i32x8 g1;
  unsigned w0 = (1u << 16);                              // data_size=1 (2B)
  if (pad) w0 |= (1u << 20) | (6u << 22) | (3u << 25);   // pad 4dw every 128dw
  g1[0] = (int)w0;
  g1[1] = (int)((d0 & 0xFFFFu) << 16);                   // tensor_dim0 lo
  g1[2] = (int)(((d0 >> 16) & 0xFFFFu) | ((d1 & 0xFFFFu) << 16));
  g1[3] = (int)(((d1 >> 16) & 0xFFFFu) | ((d0 & 0xFFFFu) << 16)); // tile_dim0
  g1[4] = (int)(d1 & 0xFFFFu);                           // tile_dim1, dim2=0
  g1[5] = (int)stride;                                   // dim0 stride lo32
  g1[6] = 0;
  g1[7] = 0;
  i32x4 z4 = {0, 0, 0, 0};
  i32x8 z8 = {0, 0, 0, 0, 0, 0, 0, 0};
  // clang-23 / therock form: 6 args (g0, g1, g2, g3, unused-group, cpol)
  __builtin_amdgcn_tensor_load_to_lds(g0, g1, z4, z4, z8, 0);
}
#endif

__device__ __forceinline__ void tdm_wait() {
#if HAVE_TDM && __has_builtin(__builtin_amdgcn_s_wait_tensorcnt)
  __builtin_amdgcn_s_wait_tensorcnt(0);
#endif
}

// Stage one 16-row x 256-f16 B tile (contiguous 8KB in xh) into LDS with the
// padded BROW_BYTES layout. TDM path: single instruction from one lane.
__device__ __forceinline__ void stage_b_tile(char* dst, const _Float16* src,
                                             int tid) {
#if HAVE_TDM
  if (tid == 0)
    tdm_load_2d_f16((unsigned)(unsigned long long)(void*)dst, src,
                    CIN, 16, CIN, /*pad=*/true);
#else
  const v8h* s = (const v8h*)src;                        // 512 x 16B chunks
  for (int c = tid; c < 512; c += 128) {
    const int r = c >> 5, cc = c & 31;
    *(v8h*)(dst + r * BROW_BYTES + cc * 16) = s[r * 32 + cc];
  }
#endif
}

// ---------------------------------------------------------------------------
// prep_x: x[b][c][n] (f32) -> xh[b][n][c] (f16), sq[b][n] = sum_c x^2
// ---------------------------------------------------------------------------
__global__ __launch_bounds__(256) void prep_x_kernel(const float* __restrict__ x,
                                                     _Float16* __restrict__ xh,
                                                     float* __restrict__ sq) {
  __shared__ float red[8][32];
  const int b  = blockIdx.x >> 7;              // NPT/32 = 128 tiles per batch
  const int n0 = (blockIdx.x & 127) * 32;
  const int tx = threadIdx.x, ty = threadIdx.y;
  const int n  = n0 + tx;
  const float* xb = x + (size_t)b * CIN * NPT;
  _Float16* xr = xh + ((size_t)b * NPT + n) * CIN;
  float s = 0.f;
  for (int c = ty; c < CIN; c += 8) {
    float v = xb[(size_t)c * NPT + n];         // coalesced over tx
    xr[c] = (_Float16)v;
    s += v * v;
  }
  red[ty][tx] = s;
  __syncthreads();
  if (ty == 0) {
    float t = 0.f;
#pragma unroll
    for (int q = 0; q < 8; ++q) t += red[q][tx];
    sq[(size_t)b * NPT + n] = t;
  }
}

// ---------------------------------------------------------------------------
// prep_w: W[o][c][k] (f32) -> Wh[o][k*256+c] (f16)
// ---------------------------------------------------------------------------
__global__ __launch_bounds__(256) void prep_w_kernel(const float* __restrict__ W,
                                                     _Float16* __restrict__ Wh) {
  const int t = blockIdx.x * 256 + threadIdx.x;
  const int o  = t / KK_TOT;
  const int kk = t % KK_TOT;
  const int k  = kk >> 8;
  const int c  = kk & 255;
  Wh[t] = (_Float16)W[((size_t)o * CIN + c) * KNN + k];
}

// ---------------------------------------------------------------------------
// knn_topk: fused distance-GEMM + running top-3.
// grid = BATCH*(NPT/64), block = 128 (4 waves, each owns a 16-row A panel).
// B tiles staged once per WG via TDM (double buffered), shared by all 4 waves.
// Ordering key for row i, col j: sq[j] - 2*G[i][j] (== dist up to +sq_i).
// ---------------------------------------------------------------------------
__global__ __launch_bounds__(128) void knn_topk_kernel(
    const _Float16* __restrict__ xh, const float* __restrict__ sq,
    int* __restrict__ idx) {
  __shared__ __align__(16) char Bsh[2][16 * BROW_BYTES];  // 2 x 8.25KB
  __shared__ float keys[4][16][17];
  __shared__ float fd[4][16][3];
  __shared__ int   fi[4][16][3];
  const int tid  = threadIdx.x;
  const int wave = tid >> 5;
  const int lane = tid & 31;
  const int b    = blockIdx.x >> 6;            // 64 row-groups per batch
  const int i0   = ((blockIdx.x & 63) * 4 + wave) * 16;
  const _Float16* xb  = xh + (size_t)b * NPT * CIN;
  const float*    sqb = sq + (size_t)b * NPT;

  // Stage tile 0 (TDM overlaps the A-panel preload below)
  stage_b_tile(Bsh[0], xb, tid);

  // Preload A panel: rows i0..i0+15, K = 0..255 (8 chunks of 32)
  v16h A[8];
#pragma unroll
  for (int c8 = 0; c8 < 8; ++c8)
    A[c8] = load_a_chunk(xb + (size_t)i0 * CIN, CIN, c8 * 32, lane);

  tdm_wait();
  __syncthreads();

  float best0 = INFINITY, best1 = INFINITY, best2 = INFINITY;
  int   bi0 = 0, bi1 = 0, bi2 = 0;
  const int col   = lane & 15;
  const int rbase = (lane < 16) ? 0 : 8;       // C/D row group
  const int cb    = (lane < 16) ? 0 : 8;       // merge column base

  for (int jt = 0; jt < NPT / 16; ++jt) {
    const int j0  = jt * 16;
    const int buf = jt & 1;
    // Kick off TDM for the next tile into the other buffer (overlaps compute)
    if (jt + 1 < NPT / 16)
      stage_b_tile(Bsh[buf ^ 1], xb + (size_t)(j0 + 16) * CIN, tid);

    const float sqj = sqb[j0 + col];
    const _Float16* colrow = (const _Float16*)(Bsh[buf] + col * BROW_BYTES);

    // Depth-1 software pipeline: loads for chunk c8+1 issue before WMMA c8,
    // so each chunk's ds_load latency hides under the previous WMMA.
    v8f acc = {};
    v16h Bcur = load_b_chunk(colrow, 0, lane);
#pragma unroll
    for (int c8 = 0; c8 < 8; ++c8) {
      v16h Bnxt = Bcur;
      if (c8 + 1 < 8) Bnxt = load_b_chunk(colrow, (c8 + 1) * 32, lane);
      acc = wmma_f16(A[c8], Bcur, acc);
      Bcur = Bnxt;
    }

    // Scatter ordering keys into this wave's private LDS slice
#pragma unroll
    for (int v = 0; v < 8; ++v)
      keys[wave][rbase + v][col] = fmaf(-2.0f, acc[v], sqj);
    // DS ops are in-order per wave: compiler fence is sufficient, no barrier
    asm volatile("" ::: "memory");

    // All-lane merge: lane handles row (lane&15), columns cb..cb+7.
    // Outer guard usually fails on all lanes (execz skip); body is branchless.
#pragma unroll
    for (int t = 0; t < 8; ++t) {
      const float d = keys[wave][col][cb + t];
      const int   j = j0 + cb + t;
      if (d < best2) {
        const bool lt1 = d < best1;
        const bool lt0 = d < best0;
        best2 = lt1 ? best1 : d;
        bi2   = lt1 ? bi1   : j;
        const float nb1 = lt0 ? best0 : (lt1 ? d : best1);
        const int   ni1 = lt0 ? bi0   : (lt1 ? j : bi1);
        best1 = nb1;  bi1 = ni1;
        best0 = lt0 ? d : best0;
        bi0   = lt0 ? j : bi0;
      }
    }

    tdm_wait();          // next-tile TDM complete (issuing wave; no-op others)
    __syncthreads();     // all waves done reading buf before it is overwritten
  }

  // Cross-half merge: lanes >=16 publish their sorted top-3, lanes <16 merge
  if (lane >= 16) {
    fd[wave][col][0] = best0; fd[wave][col][1] = best1; fd[wave][col][2] = best2;
    fi[wave][col][0] = bi0;   fi[wave][col][1] = bi1;   fi[wave][col][2] = bi2;
  }
  asm volatile("" ::: "memory");
  if (lane < 16) {
    const float av1 = best1, av2 = best2;
    const int   ai1 = bi1,   ai2 = bi2;
    const float bv0 = fd[wave][lane][0], bv1 = fd[wave][lane][1],
                bv2 = fd[wave][lane][2];
    const int   bj0 = fi[wave][lane][0], bj1 = fi[wave][lane][1],
                bj2 = fi[wave][lane][2];
    float aH = best0, bH = bv0;
    int   aI = bi0,   bI = bj0;
    int   ap = 0,     bp = 0;
    int   rj[3];
#pragma unroll
    for (int t = 0; t < 3; ++t) {
      const bool ta = aH <= bH;
      rj[t] = ta ? aI : bI;
      if (ta) { ++ap; aH = (ap == 1) ? av1 : av2; aI = (ap == 1) ? ai1 : ai2; }
      else    { ++bp; bH = (bp == 1) ? bv1 : bv2; bI = (bp == 1) ? bj1 : bj2; }
    }
    int* op = idx + ((size_t)b * NPT + i0 + lane) * KNN;
    op[0] = rj[0]; op[1] = rj[1]; op[2] = rj[2];   // nearest-first
  }
}

// ---------------------------------------------------------------------------
// conv_nn: out[b,o,n] = bias[o] + sum_kk Wh[o][kk] * xh[b][idx[b,n,k]][c]
// grid = BATCH*16*64, block = 128; wave handles one 16(o) x 16(n) tile.
// Flattened 24-chunk K loop with depth-2 software pipeline over global loads.
// ---------------------------------------------------------------------------
__global__ __launch_bounds__(128) void conv_nn_kernel(
    const _Float16* __restrict__ xh, const _Float16* __restrict__ Wh,
    const float* __restrict__ bias, const int* __restrict__ idx,
    float* __restrict__ out) {
  const int wave = threadIdx.x >> 5;
  const int lane = threadIdx.x & 31;
  const int bid  = blockIdx.x;
  const int b    = bid >> 10;                   // 16 otiles * 64 ngroups
  const int rem  = bid & 1023;
  const int o0   = (rem >> 6) * 16;
  const int n0   = ((rem & 63) * 4 + wave) * 16;
  const int col  = lane & 15;
  const int n    = n0 + col;

  // Gather: 3 neighbor row pointers for this lane's column
  const int* ip = idx + ((size_t)b * NPT + n) * KNN;
  const _Float16* xb = xh + (size_t)b * NPT * CIN;
  const _Float16* crow0 = xb + (size_t)ip[0] * CIN;
  const _Float16* crow1 = xb + (size_t)ip[1] * CIN;
  const _Float16* crow2 = xb + (size_t)ip[2] * CIN;
  const _Float16* crow[3] = { crow0, crow1, crow2 };

  const _Float16* arow = Wh + (size_t)o0 * KK_TOT;
  v8f acc = {};
  // Chunk q (0..23): k = q>>3, c8 = q&7; kk offset = k*CIN + c8*32
  v16h Ab[2], Bb[2];
  Ab[0] = load_a_chunk(arow, KK_TOT, 0, lane);
  Bb[0] = load_b_chunk(crow[0], 0, lane);
  Ab[1] = load_a_chunk(arow, KK_TOT, 32, lane);
  Bb[1] = load_b_chunk(crow[0], 32, lane);
#pragma unroll
  for (int q = 0; q < 24; ++q) {
    const int cur = q & 1;
    acc = wmma_f16(Ab[cur], Bb[cur], acc);
    if (q + 2 < 24) {
      const int qq = q + 2;
      const int k  = qq >> 3, c8 = qq & 7;
      Ab[cur] = load_a_chunk(arow, KK_TOT, k * CIN + c8 * 32, lane);
      Bb[cur] = load_b_chunk(crow[k], c8 * 32, lane);
    }
  }

  const int rbase = (lane < 16) ? 0 : 8;
#pragma unroll
  for (int v = 0; v < 8; ++v) {
    const int o = o0 + rbase + v;
    out[((size_t)b * COUT + o) * NPT + n] = acc[v] + bias[o];
  }
}

// ---------------------------------------------------------------------------
extern "C" void kernel_launch(void* const* d_in, const int* in_sizes, int n_in,
                              void* d_out, int out_size, void* d_ws, size_t ws_size,
                              hipStream_t stream) {
  (void)in_sizes; (void)n_in; (void)out_size; (void)ws_size;
  const float* x    = (const float*)d_in[0];   // [B, CIN, N] f32
  const float* W    = (const float*)d_in[1];   // [COUT, CIN, K] f32
  const float* bias = (const float*)d_in[2];   // [COUT] f32
  float* out = (float*)d_out;                  // [B, COUT, N] f32

  char* ws = (char*)d_ws;
  _Float16* xh  = (_Float16*)(ws);
  float*    sq  = (float*)   (ws + XH_BYTES);
  _Float16* Wh  = (_Float16*)(ws + XH_BYTES + SQ_BYTES);
  int*      idx = (int*)     (ws + XH_BYTES + SQ_BYTES + WH_BYTES);

  prep_x_kernel<<<dim3(BATCH * (NPT / 32)), dim3(32, 8), 0, stream>>>(x, xh, sq);
  prep_w_kernel<<<dim3((COUT * KK_TOT) / 256), dim3(256), 0, stream>>>(W, Wh);
  knn_topk_kernel<<<dim3(BATCH * (NPT / 64)), dim3(128), 0, stream>>>(xh, sq, idx);
  conv_nn_kernel<<<dim3(BATCH * 16 * 64), dim3(128), 0, stream>>>(xh, Wh, bias, idx, out);
}
